// CLG_64785286693460
// MI455X (gfx1250) — compile-verified, matrix operand order checked by reading, not run on hardware
//
#include <hip/hip_runtime.h>
#include <hip/hip_bf16.h>
#include <stdint.h>

// GCN 2-layer forward for MI455X (gfx1250, wave32).
// GEMMs via v_wmma_f32_16x16x32_f16 (f16 in, f32 accum) with b128 fragment
// loads (weights pre-transposed); edge aggregation via L2-resident float
// atomics (H1/agg1 ~51MB each, fit in 192MB L2).

typedef __attribute__((ext_vector_type(16))) _Float16 v16h;
typedef __attribute__((ext_vector_type(8)))  _Float16 h8;
typedef __attribute__((ext_vector_type(8)))  float    v8f;

#define GCN_DIN 128
#define GCN_DH  128
#define GCN_DOUT 16

// ---------------- elementwise helpers ----------------

__global__ void k_cvt_f32_f16(const float* __restrict__ in,
                              _Float16* __restrict__ out, size_t n) {
  size_t i = (size_t)blockIdx.x * blockDim.x + threadIdx.x;
  if (i < n) out[i] = (_Float16)in[i];
}

// in: R x C row-major f32  ->  out: C x R row-major f16 (transpose + convert)
__global__ void k_cvt_t_f32_f16(const float* __restrict__ in,
                                _Float16* __restrict__ out, int R, int C) {
  int i = blockIdx.x * blockDim.x + threadIdx.x;
  if (i < R * C) {
    int r = i / C, c = i % C;
    out[(size_t)c * R + r] = (_Float16)in[i];
  }
}

__global__ void k_init_deg(float* __restrict__ deg, int n) {
  int i = blockIdx.x * blockDim.x + threadIdx.x;
  if (i < n) deg[i] = 1.0f;  // self-loop contributes 1 to every node's degree
}

__global__ void k_accum_deg(const long long* __restrict__ dst,
                            float* __restrict__ deg, int e) {
  int i = blockIdx.x * blockDim.x + threadIdx.x;
  if (i < e) atomicAdd(&deg[(int)dst[i]], 1.0f);
}

__global__ void k_deg_to_dinv(float* __restrict__ deg, int n) {
  int i = blockIdx.x * blockDim.x + threadIdx.x;
  if (i < n) {
    float d = deg[i];
    deg[i] = (d > 0.0f) ? rsqrtf(d) : 0.0f;
  }
}

// ---------------- WMMA GEMM: C[M x NC] = A[M x K] * Bt[NC x K]^T -------------
// One wave per 16x16 output tile. f16 A/B, f32 accumulate.
// 16-bit fragment layout per lane (A and B identical, B indexed by column):
//   halves 0..7  = K in [base, base+8)   (VGPRs 0-3)
//   halves 8..15 = K in [base+16, base+24)
// with base = k0 + 8*(lane>=16); both runs are contiguous 16B in memory when
// the matrix is K-major per row -> two global_load_b128 per fragment.
// C/D: VGPR j -> row (j + 8*(lane>=16)), col = lane&15.

template <int K, int NC>
__global__ void k_gemm_wmma_f16(const _Float16* __restrict__ A,
                                const _Float16* __restrict__ Bt,
                                float* __restrict__ C, int M) {
  const int TILES_N = NC / 16;
  int wave = (int)(((size_t)blockIdx.x * blockDim.x + threadIdx.x) >> 5);
  int lane = threadIdx.x & 31;
  int tm = (wave / TILES_N) * 16;
  int tn = (wave % TILES_N) * 16;
  if (tm >= M) return;

  const int half_id = lane >> 4;
  const int l15 = lane & 15;
  const _Float16* __restrict__ arow = A + (size_t)(tm + l15) * K + (half_id << 3);
  const _Float16* __restrict__ brow = Bt + (size_t)(tn + l15) * K + (half_id << 3);

  v8f acc = {};
#pragma unroll
  for (int k0 = 0; k0 < K; k0 += 32) {
    h8 alo = *(const h8*)(arow + k0);
    h8 ahi = *(const h8*)(arow + k0 + 16);
    h8 blo = *(const h8*)(brow + k0);
    h8 bhi = *(const h8*)(brow + k0 + 16);
    v16h a = __builtin_shufflevector(alo, ahi, 0, 1, 2, 3, 4, 5, 6, 7,
                                     8, 9, 10, 11, 12, 13, 14, 15);
    v16h b = __builtin_shufflevector(blo, bhi, 0, 1, 2, 3, 4, 5, 6, 7,
                                     8, 9, 10, 11, 12, 13, 14, 15);
    acc = __builtin_amdgcn_wmma_f32_16x16x32_f16(
        /*neg_a=*/false, a, /*neg_b=*/false, b,
        /*c_mod=*/(short)0, acc, /*reuse_a=*/false, /*reuse_b=*/false);
  }

#pragma unroll
  for (int j = 0; j < 8; ++j) {
    C[(size_t)(tm + (half_id << 3) + j) * NC + tn + l15] = acc[j];
  }
}

// ---------------- aggregation (scatter-add over edges) ----------------

// Initialize agg1 with the self-loop term: agg[i][f] = H[i][f] * dinv[i]^2
__global__ void k_selfloop128(const float* __restrict__ H,
                              const float* __restrict__ dinv,
                              float* __restrict__ out, size_t n) {
  size_t i = (size_t)blockIdx.x * blockDim.x + threadIdx.x;
  if (i < n) {
    float di = dinv[i >> 7];
    out[i] = H[i] * di * di;
  }
}

// One wave per edge; lane handles 4 consecutive features (float4 gather, L2 atomics).
__global__ void k_agg128_edges(const long long* __restrict__ src,
                               const long long* __restrict__ dst,
                               const float* __restrict__ dinv,
                               const float* __restrict__ H,
                               float* __restrict__ out, int e) {
  int edge = (int)(((size_t)blockIdx.x * blockDim.x + threadIdx.x) >> 5);
  int lane = threadIdx.x & 31;
  if (edge >= e) return;
  int s = (int)src[edge], d = (int)dst[edge];
  float nrm = dinv[s] * dinv[d];
  float4 v = ((const float4*)(H + (size_t)s * GCN_DH))[lane];
  float* orow = out + (size_t)d * GCN_DH + lane * 4;
  atomicAdd(orow + 0, v.x * nrm);
  atomicAdd(orow + 1, v.y * nrm);
  atomicAdd(orow + 2, v.z * nrm);
  atomicAdd(orow + 3, v.w * nrm);
}

// relu(agg + b1) converted straight to f16 for GEMM2
__global__ void k_relu_bias_cvt(const float* __restrict__ agg,
                                const float* __restrict__ b,
                                _Float16* __restrict__ outh, size_t n) {
  size_t i = (size_t)blockIdx.x * blockDim.x + threadIdx.x;
  if (i < n) {
    float v = agg[i] + b[i & (GCN_DH - 1)];
    outh[i] = (_Float16)(v > 0.0f ? v : 0.0f);
  }
}

// out init with self-loop + bias: out[i][f] = H2[i][f]*dinv[i]^2 + b2[f]
__global__ void k_selfloop16_bias(const float* __restrict__ H2,
                                  const float* __restrict__ dinv,
                                  const float* __restrict__ b2,
                                  float* __restrict__ out, size_t n) {
  size_t i = (size_t)blockIdx.x * blockDim.x + threadIdx.x;
  if (i < n) {
    float di = dinv[i >> 4];
    out[i] = H2[i] * di * di + b2[i & (GCN_DOUT - 1)];
  }
}

// 16 threads per edge (one feature each)
__global__ void k_agg16_edges(const long long* __restrict__ src,
                              const long long* __restrict__ dst,
                              const float* __restrict__ dinv,
                              const float* __restrict__ H2,
                              float* __restrict__ out, size_t n /* e*16 */) {
  size_t t = (size_t)blockIdx.x * blockDim.x + threadIdx.x;
  if (t >= n) return;
  size_t edge = t >> 4;
  int f = (int)(t & 15);
  int s = (int)src[edge], d = (int)dst[edge];
  float nrm = dinv[s] * dinv[d];
  atomicAdd(&out[(size_t)d * GCN_DOUT + f], H2[(size_t)s * GCN_DOUT + f] * nrm);
}

// ---------------- launcher ----------------

extern "C" void kernel_launch(void* const* d_in, const int* in_sizes, int n_in,
                              void* d_out, int out_size, void* d_ws, size_t ws_size,
                              hipStream_t stream) {
  const float*     x   = (const float*)d_in[0];          // [N, 128]
  const long long* ei  = (const long long*)d_in[1];      // [2, E] int64
  const float*     W1  = (const float*)d_in[2];          // [128, 128]
  const float*     b1  = (const float*)d_in[3];          // [128]
  const float*     W2  = (const float*)d_in[4];          // [128, 16]
  const float*     b2  = (const float*)d_in[5];          // [16]
  float*           out = (float*)d_out;                  // [N, 16]

  const int N = in_sizes[0] / GCN_DIN;
  const int E = in_sizes[1] / 2;
  const long long* src = ei;
  const long long* dst = ei + E;

  // workspace carve-up (256B aligned)
  size_t off = 0;
  auto carve = [&](size_t bytes) -> char* {
    char* p = (char*)d_ws + off;
    off = (off + bytes + 255) & ~(size_t)255;
    return p;
  };
  float*    dinv = (float*)carve((size_t)N * sizeof(float));
  _Float16* Xh   = (_Float16*)carve((size_t)N * GCN_DIN * sizeof(_Float16)); // reused as A1h
  _Float16* W1t  = (_Float16*)carve((size_t)GCN_DIN * GCN_DH * sizeof(_Float16));  // [DH, DIN]
  _Float16* W2t  = (_Float16*)carve((size_t)GCN_DH * GCN_DOUT * sizeof(_Float16)); // [DOUT, DH]
  float*    H1   = (float*)carve((size_t)N * GCN_DH * sizeof(float));        // reused as H2
  float*    agg1 = (float*)carve((size_t)N * GCN_DH * sizeof(float));
  if (off > ws_size) return;  // workspace too small; bail safely

  const int BT = 256;
  auto blocks = [&](size_t n) -> unsigned { return (unsigned)((n + BT - 1) / BT); };

  // 1) f16 conversions (weights transposed so WMMA B-fragments are contiguous)
  k_cvt_f32_f16<<<blocks((size_t)N * GCN_DIN), BT, 0, stream>>>(x, Xh, (size_t)N * GCN_DIN);
  k_cvt_t_f32_f16<<<blocks(GCN_DIN * GCN_DH), BT, 0, stream>>>(W1, W1t, GCN_DIN, GCN_DH);
  k_cvt_t_f32_f16<<<blocks(GCN_DH * GCN_DOUT), BT, 0, stream>>>(W2, W2t, GCN_DH, GCN_DOUT);

  // 2) degrees -> dinv (in place)
  k_init_deg<<<blocks(N), BT, 0, stream>>>(dinv, N);
  k_accum_deg<<<blocks(E), BT, 0, stream>>>(dst, dinv, E);
  k_deg_to_dinv<<<blocks(N), BT, 0, stream>>>(dinv, N);

  // 3) H1 = X @ W1  (WMMA; 8 N-tiles per row-tile, 8 waves/block)
  {
    size_t waves = ((size_t)N / 16) * (GCN_DH / 16);
    k_gemm_wmma_f16<GCN_DIN, GCN_DH><<<blocks(waves * 32), BT, 0, stream>>>(Xh, W1t, H1, N);
  }

  // 4) agg1 = Â · H1 (self-loop init, then edge scatter; one wave per edge)
  k_selfloop128<<<blocks((size_t)N * GCN_DH), BT, 0, stream>>>(H1, dinv, agg1, (size_t)N * GCN_DH);
  k_agg128_edges<<<blocks((size_t)E * 32), BT, 0, stream>>>(src, dst, dinv, H1, agg1, E);

  // 5) A1h = f16(relu(agg1 + b1))  (into Xh buffer, no longer needed)
  k_relu_bias_cvt<<<blocks((size_t)N * GCN_DH), BT, 0, stream>>>(agg1, b1, Xh, (size_t)N * GCN_DH);

  // 6) H2 = A1 @ W2  (WMMA, single N-tile; into H1 buffer)
  {
    size_t waves = (size_t)N / 16;  // TILES_N = 1
    k_gemm_wmma_f16<GCN_DH, GCN_DOUT><<<blocks(waves * 32), BT, 0, stream>>>(Xh, W2t, H1, N);
  }

  // 7) out = Â · H2 + b2
  k_selfloop16_bias<<<blocks((size_t)N * GCN_DOUT), BT, 0, stream>>>(H1, dinv, b2, out, (size_t)N * GCN_DOUT);
  k_agg16_edges<<<blocks((size_t)E * GCN_DOUT), BT, 0, stream>>>(src, dst, dinv, H1, out, (size_t)E * GCN_DOUT);
}
